// FactoredPreLNBlock_44135083934210
// MI455X (gfx1250) — compile-verified, matrix operand order checked by reading, not run on hardware
//
#include <hip/hip_runtime.h>
#include <hip/hip_bf16.h>
#include <stdint.h>

// ---------------------------------------------------------------------------
// FactoredPreLNBlock for MI455X (gfx1250, wave32, WMMA).
// GEMMs + both attention matmuls: v_wmma_f32_16x16x32_bf16.
// GEMM staging: TDM tensor_load_to_lds (pad-enabled) w/ s_wait_tensorcnt,
// fallback global_load_async_to_lds_b128 + s_wait_asynccnt. Double buffered.
// ---------------------------------------------------------------------------

#define N_EMBD 1024
#define N_HEADS 16
#define HEAD_D 64
#define BATCH 4
#define SEQ 2048
#define MTOK (BATCH * SEQ)        // 8192 rows
#define C3 (3 * N_EMBD)           // 3072
#define C4 (4 * N_EMBD)           // 4096

#if defined(__has_builtin)
#if __has_builtin(__builtin_amdgcn_tensor_load_to_lds) && \
    __has_builtin(__builtin_amdgcn_s_wait_tensorcnt)
#define HAVE_TDM 1
#endif
#endif
#ifndef HAVE_TDM
#define HAVE_TDM 0
#endif

typedef __attribute__((ext_vector_type(16))) __bf16 v16bf;
typedef __attribute__((ext_vector_type(8)))  float  v8f;
typedef __attribute__((ext_vector_type(4))) unsigned int v4u;
typedef __attribute__((ext_vector_type(8))) int v8i;
typedef __attribute__((ext_vector_type(4))) int v4i;

// Native bf16 <-> f32.
__device__ __forceinline__ uint16_t f2bf(float f) {
  __bf16 h = (__bf16)f;
  uint16_t u;
  __builtin_memcpy(&u, &h, 2);
  return u;
}
__device__ __forceinline__ float bf2f(uint16_t x) {
  __bf16 h;
  __builtin_memcpy(&h, &x, 2);
  return (float)h;
}
__device__ __forceinline__ uint32_t pk2bf(float lo, float hi) {
  return (uint32_t)f2bf(lo) | ((uint32_t)f2bf(hi) << 16);
}

struct Frag {
  union {
    uint32_t u[8];
    v16bf    v;
  };
};

__device__ __forceinline__ v8f fzero() {
  v8f z = {0.f, 0.f, 0.f, 0.f, 0.f, 0.f, 0.f, 0.f};
  return z;
}

__device__ __forceinline__ v8f wmma_bf16(const Frag& a, const Frag& b, v8f c) {
  return __builtin_amdgcn_wmma_f32_16x16x32_bf16(
      false, a.v, false, b.v, (short)0, c, false, false);
}

// Low 32 bits of a flat shared address == LDS byte offset (ISA 10.2).
__device__ __forceinline__ uint32_t lds_off(const void* p) {
  return (uint32_t)(uintptr_t)p;
}

// Async 16-byte copy global -> LDS (per lane). Tracked by ASYNCcnt.
__device__ __forceinline__ void async_cp16(uint32_t ldsOff, uint32_t gOff,
                                           const void* base) {
  asm volatile("global_load_async_to_lds_b128 %0, %1, %2"
               :: "v"(ldsOff), "v"(gOff), "s"(base)
               : "memory");
}
__device__ __forceinline__ void wait_async0() {
  asm volatile("s_wait_asynccnt 0x0" ::: "memory");
}

#if HAVE_TDM
// TDM: DMA a rows x 32-elem bf16 tile (row stride K elems in memory) into LDS
// with 8-DWORD padding after every 16 DWORDs -> 96-byte LDS rows (LDS_S=48).
__device__ __forceinline__ void tdm_load_tile(uint32_t lds_byte,
                                              const void* gbase,
                                              uint64_t elem_off, int Kelems,
                                              int rows) {
  const uint64_t ga = (uint64_t)gbase + elem_off * 2;
  v4u g0;
  g0[0] = 1u;                                            // count=1 (valid)
  g0[1] = lds_byte;                                      // lds_addr
  g0[2] = (uint32_t)ga;                                  // global_addr[31:0]
  g0[3] = (uint32_t)((ga >> 32) & 0x1FFFFFFu) | (2u << 30);  // [56:32] | type=2
  v8i g1;
  // data_size=2B | pad_enable | pad_interval=3 (16 DW) | pad_amount=7 (8 DW)
  g1[0] = (int)((1u << 16) | (1u << 20) | (3u << 22) | (7u << 25));
  g1[1] = (int)(((uint32_t)Kelems & 0xFFFFu) << 16);     // tensor_dim0 lo
  g1[2] = (int)(((uint32_t)Kelems >> 16) & 0xFFFFu);     // tensor_dim0 hi | dim1 lo=0
  g1[3] = (int)(0x10u | (32u << 16));                    // dim1 hi (=1M rows) | tile_dim0=32
  g1[4] = (int)(uint32_t)rows;                           // tile_dim1 | tile_dim2=0
  g1[5] = (int)(uint32_t)Kelems;                         // dim0_stride[31:0]
  g1[6] = 0;                                             // dim0_stride hi | dim1_stride lo
  g1[7] = 0;
  v4i gz4 = {0, 0, 0, 0};
  v8i gz8 = {0, 0, 0, 0, 0, 0, 0, 0};
  // clang-23 / therock form: 6 args (g0, g1, g2, g3, <v8i>, cpol).
  __builtin_amdgcn_tensor_load_to_lds(g0, g1, gz4, gz4, gz8, 0);
}
#endif

// Load a 16x32 bf16 fragment (A layout; same pattern serves B when the tile
// is stored [N][K] row-major) from LDS.
__device__ __forceinline__ void lds_frag(const uint16_t* tile, int row0,
                                         int stride, int kbase, Frag& f) {
  const int lane = threadIdx.x & 31;
  const int m  = lane & 15;
  const int hi = lane >> 4;
#pragma unroll
  for (int vv = 0; vv < 8; ++vv) {
    const int k = kbase + ((vv & 4) << 2) + (hi << 3) + ((vv & 3) << 1);
    f.u[vv] = *(const uint32_t*)(tile + (size_t)(row0 + m) * stride + k);
  }
}

// ---------------------------------------------------------------------------
// f32 -> bf16 conversion (weights)
// ---------------------------------------------------------------------------
__global__ __launch_bounds__(256) void k_f32_to_bf16(const float* __restrict__ src,
                                                     uint16_t* __restrict__ dst,
                                                     int n) {
  int i = blockIdx.x * 256 + threadIdx.x;
  if (i < n) dst[i] = f2bf(src[i]);
}

// ---------------------------------------------------------------------------
// Fused residual-add + LayerNorm -> bf16 (one block per token row of 1024)
// ---------------------------------------------------------------------------
__device__ __forceinline__ float block_sum(float s, float* red) {
#pragma unroll
  for (int o = 16; o > 0; o >>= 1) s += __shfl_down(s, o, 32);
  const int w = threadIdx.x >> 5;
  if ((threadIdx.x & 31) == 0) red[w] = s;
  __syncthreads();
  if (threadIdx.x < 32) {
    float t = (threadIdx.x < 8) ? red[threadIdx.x] : 0.f;
#pragma unroll
    for (int o = 4; o > 0; o >>= 1) t += __shfl_down(t, o, 32);
    if (threadIdx.x == 0) red[0] = t;
  }
  __syncthreads();
  float r = red[0];
  __syncthreads();
  return r;
}

__global__ __launch_bounds__(256) void k_add_ln_bf16(
    const float* __restrict__ x1, const float* __restrict__ x2,
    const float* __restrict__ gw, const float* __restrict__ gb,
    uint16_t* __restrict__ out, uint16_t* __restrict__ x1b) {
  __shared__ float red[8];
  const int row = blockIdx.x;
  const int tid = threadIdx.x;
  const size_t base = (size_t)row * N_EMBD;
  float v[4];
  float s = 0.f;
#pragma unroll
  for (int i = 0; i < 4; ++i) {
    const int c = tid + i * 256;
    const float a = x1[base + c];
    const float b = x2[base + c];
    if (x1b) x1b[base + c] = f2bf(a);
    v[i] = a + b;
    s += v[i];
  }
  const float mean = block_sum(s, red) * (1.f / N_EMBD);
  float ss = 0.f;
#pragma unroll
  for (int i = 0; i < 4; ++i) {
    const float d = v[i] - mean;
    ss += d * d;
  }
  const float rstd = rsqrtf(block_sum(ss, red) * (1.f / N_EMBD) + 1e-5f);
#pragma unroll
  for (int i = 0; i < 4; ++i) {
    const int c = tid + i * 256;
    out[base + c] = f2bf((v[i] - mean) * rstd * gw[c] + gb[c]);
  }
}

// ---------------------------------------------------------------------------
// Tiled bf16 WMMA GEMM:  out[M,N] = A[M,K] @ B[N,K]^T  (+bias, epilogue)
// 128 threads = 4 waves; workgroup tile 128x128; wave tile 64x64; K-step 32.
// Double-buffered LDS staged by TDM (or async fallback), overlapping 16 WMMAs.
// EPI: 0 = bias -> bf16 ; 1 = bias + exact GELU -> bf16 ; 2 = bias + resid -> f32
// ---------------------------------------------------------------------------
#define LDS_S 48  // padded LDS row stride (bf16 elems): 96B rows

template <int EPI>
__global__ __launch_bounds__(128) void k_gemm_bf16(
    const uint16_t* __restrict__ A, const uint16_t* __restrict__ Bm,
    const float* __restrict__ bias, const float* __restrict__ resid,
    void* __restrict__ out, int M, int N, int K) {
  __shared__ uint16_t lA[2][128 * LDS_S];
  __shared__ uint16_t lB[2][128 * LDS_S];

  const int m0 = blockIdx.x * 128;
  const int n0 = blockIdx.y * 128;
  const int tid = threadIdx.x;
  const int w = tid >> 5;
  const int wm = w & 1;   // 2 waves along M (64 rows each)
  const int wn = w >> 1;  // 2 waves along N (64 cols each)

  // Issue staging for K-tile at k0 into buffer `buf`.
  auto stage = [&](int k0, int buf) {
#if HAVE_TDM
    if (w == 0) {
      tdm_load_tile(lds_off(&lA[buf][0]), A, (uint64_t)m0 * K + k0, K, 128);
      tdm_load_tile(lds_off(&lB[buf][0]), Bm, (uint64_t)n0 * K + k0, K, 128);
    }
#else
#pragma unroll
    for (int i = 0; i < 4; ++i) {
      const int c = tid + i * 128;         // 0..511 chunks of 8 bf16
      const int r = c >> 2;
      const int col = (c & 3) * 8;
      async_cp16(lds_off(&lA[buf][r * LDS_S + col]),
                 (uint32_t)(((size_t)(m0 + r) * K + k0 + col) * 2), A);
      async_cp16(lds_off(&lB[buf][r * LDS_S + col]),
                 (uint32_t)(((size_t)(n0 + r) * K + k0 + col) * 2), Bm);
    }
#endif
  };

  v8f acc[4][4];
#pragma unroll
  for (int i = 0; i < 4; ++i)
#pragma unroll
    for (int j = 0; j < 4; ++j) acc[i][j] = fzero();

  stage(0, 0);  // prologue

  const int niter = K / 32;
  for (int it = 0; it < niter; ++it) {
    const int buf = it & 1;
#if HAVE_TDM
    if (w == 0) __builtin_amdgcn_s_wait_tensorcnt(0);
#else
    wait_async0();
#endif
    __syncthreads();  // data landed + everyone done reading the other buffer
    if (it + 1 < niter) stage((it + 1) * 32, buf ^ 1);

    Frag af[4], bf[4];
#pragma unroll
    for (int mt = 0; mt < 4; ++mt)
      lds_frag(lA[buf], wm * 64 + mt * 16, LDS_S, 0, af[mt]);
#pragma unroll
    for (int nt = 0; nt < 4; ++nt)
      lds_frag(lB[buf], wn * 64 + nt * 16, LDS_S, 0, bf[nt]);
#pragma unroll
    for (int mt = 0; mt < 4; ++mt)
#pragma unroll
      for (int nt = 0; nt < 4; ++nt)
        acc[mt][nt] = wmma_bf16(af[mt], bf[nt], acc[mt][nt]);
  }

  // Epilogue. C/D layout: VGPR r -> row (r + 8*hi), lanes 0..15 -> col n.
  const int lane = tid & 31;
  const int n = lane & 15;
  const int hi = lane >> 4;
#pragma unroll
  for (int mt = 0; mt < 4; ++mt)
#pragma unroll
    for (int nt = 0; nt < 4; ++nt)
#pragma unroll
      for (int r = 0; r < 8; ++r) {
        const int gm = m0 + wm * 64 + mt * 16 + r + 8 * hi;
        const int gn = n0 + wn * 64 + nt * 16 + n;
        float val = acc[mt][nt][r] + bias[gn];
        if (EPI == 1) {  // exact GELU
          val = 0.5f * val * (1.f + erff(val * 0.70710678118654752f));
        }
        if (EPI == 2) {
          ((float*)out)[(size_t)gm * N + gn] =
              resid[(size_t)gm * N + gn] + val;
        } else {
          ((uint16_t*)out)[(size_t)gm * N + gn] = f2bf(val);
        }
      }
}

// ---------------------------------------------------------------------------
// Flash attention (causal, scale 1/8), V modulated elementwise by bf16(xt).
// Grid: (T/128, H, B). 8 waves; wave w owns query rows q0..q0+15.
// K tile staged with async copies; V staged transposed+modulated (packed b32).
// ---------------------------------------------------------------------------
#define FS 72  // LDS stride for 64-wide tiles (144B rows)

__global__ __launch_bounds__(256) void k_flash_attn(
    const uint16_t* __restrict__ qkv,   // [B*T, 3C] bf16
    const uint16_t* __restrict__ xtb,   // [B*T, C]  bf16
    uint16_t* __restrict__ y) {         // [B*T, C]  bf16
  __shared__ uint16_t lK[64 * FS];        // [key][d]
  __shared__ uint16_t lVT[64 * FS];       // [d][key] (transposed, modulated)
  __shared__ uint16_t lP[8][16 * FS];     // per-wave P tile [row][key]

  const int qblk = blockIdx.x;
  const int h = blockIdx.y;
  const int b = blockIdx.z;
  const int tid = threadIdx.x;
  const int w = tid >> 5;
  const int lane = tid & 31;
  const int m = lane & 15;
  const int hi = lane >> 4;
  const int q0 = qblk * 128 + w * 16;

  // Q fragments straight from global (A layout), 2 K-steps over D=64.
  Frag qf[2];
#pragma unroll
  for (int ks = 0; ks < 2; ++ks)
#pragma unroll
    for (int vv = 0; vv < 8; ++vv) {
      const int k = ks * 32 + ((vv & 4) << 2) + (hi << 3) + ((vv & 3) << 1);
      qf[ks].u[vv] = *(const uint32_t*)(
          qkv + ((size_t)(b * SEQ) + q0 + m) * C3 + h * HEAD_D + k);
    }

  v8f o[4];
#pragma unroll
  for (int nt = 0; nt < 4; ++nt) o[nt] = fzero();
  float mrow[8], lrow[8];
#pragma unroll
  for (int r = 0; r < 8; ++r) { mrow[r] = -3e38f; lrow[r] = 0.f; }

  const int nkb = qblk * 2 + 2;  // 64-key blocks up to the causal frontier
  for (int kb = 0; kb < nkb; ++kb) {
    const int kbase = kb * 64;
    __syncthreads();  // previous iteration's LDS reads are done

    // K tile: pure async copy [key][d].
#pragma unroll
    for (int i = 0; i < 2; ++i) {
      const int c = tid + i * 256;          // 0..511 chunks of 8
      const int key = c >> 3;
      const int col = (c & 7) * 8;
      const size_t rowbase =
          ((size_t)(b * SEQ) + kbase + key) * C3 + h * HEAD_D + col;
      async_cp16(lds_off(&lK[key * FS + col]),
                 (uint32_t)((rowbase + N_EMBD) * 2), qkv);
    }
    // V tile: each thread loads a key PAIR, modulates by xt, stores packed
    // dwords transposed: lVT[d][key..key+1].
    {
      const int kp = tid >> 3;              // key pair 0..31
      const int col = (tid & 7) * 8;        // 8 d-values
      const size_t r0 =
          ((size_t)(b * SEQ) + kbase + 2 * kp) * C3 + h * HEAD_D + col;
      const size_t x0 =
          ((size_t)(b * SEQ) + kbase + 2 * kp) * N_EMBD + h * HEAD_D + col;
      uint4 va = *(const uint4*)(qkv + r0 + 2 * N_EMBD);
      uint4 vb = *(const uint4*)(qkv + r0 + 2 * N_EMBD + C3);
      uint4 xa = *(const uint4*)(xtb + x0);
      uint4 xb = *(const uint4*)(xtb + x0 + N_EMBD);
      const uint16_t* vap = (const uint16_t*)&va;
      const uint16_t* vbp = (const uint16_t*)&vb;
      const uint16_t* xap = (const uint16_t*)&xa;
      const uint16_t* xbp = (const uint16_t*)&xb;
#pragma unroll
      for (int e = 0; e < 8; ++e) {
        const uint32_t pk = pk2bf(bf2f(vap[e]) * bf2f(xap[e]),
                                  bf2f(vbp[e]) * bf2f(xbp[e]));
        *(uint32_t*)(&lVT[(col + e) * FS + 2 * kp]) = pk;
      }
    }
    wait_async0();
    __syncthreads();

    // S = Q K^T  (4 tiles of 16 keys; lK serves as B: [n=key][k=d])
    v8f sacc[4];
#pragma unroll
    for (int nt = 0; nt < 4; ++nt) {
      sacc[nt] = fzero();
#pragma unroll
      for (int ks = 0; ks < 2; ++ks) {
        Frag bf;
        lds_frag(lK, nt * 16, FS, ks * 32, bf);
        sacc[nt] = wmma_bf16(qf[ks], bf, sacc[nt]);
      }
    }

    // Scale + causal mask.
#pragma unroll
    for (int nt = 0; nt < 4; ++nt)
#pragma unroll
      for (int r = 0; r < 8; ++r) {
        const int qi = q0 + r + 8 * hi;
        const int ki = kbase + nt * 16 + m;
        float s = sacc[nt][r] * 0.125f;
        if (ki > qi) s = -3e38f;
        sacc[nt][r] = s;
      }

    // Online softmax: row stats across 4 tiles + 16 lanes of the half-wave.
    float mloc[8], mnew[8], alpha[8], rsum[8];
#pragma unroll
    for (int r = 0; r < 8; ++r)
      mloc[r] = fmaxf(fmaxf(sacc[0][r], sacc[1][r]),
                      fmaxf(sacc[2][r], sacc[3][r]));
#pragma unroll
    for (int off = 1; off < 16; off <<= 1)
#pragma unroll
      for (int r = 0; r < 8; ++r)
        mloc[r] = fmaxf(mloc[r], __shfl_xor(mloc[r], off, 32));
#pragma unroll
    for (int r = 0; r < 8; ++r) {
      mnew[r] = fmaxf(mrow[r], mloc[r]);
      alpha[r] = __expf(mrow[r] - mnew[r]);
      rsum[r] = 0.f;
    }
#pragma unroll
    for (int nt = 0; nt < 4; ++nt)
#pragma unroll
      for (int r = 0; r < 8; ++r) {
        const float p = __expf(sacc[nt][r] - mnew[r]);
        rsum[r] += p;
        lP[w][(r + 8 * hi) * FS + nt * 16 + m] = f2bf(p);
      }
#pragma unroll
    for (int off = 1; off < 16; off <<= 1)
#pragma unroll
      for (int r = 0; r < 8; ++r)
        rsum[r] += __shfl_xor(rsum[r], off, 32);
#pragma unroll
    for (int r = 0; r < 8; ++r) {
      lrow[r] = lrow[r] * alpha[r] + rsum[r];
      mrow[r] = mnew[r];
    }
#pragma unroll
    for (int nt = 0; nt < 4; ++nt)
#pragma unroll
      for (int r = 0; r < 8; ++r) o[nt][r] *= alpha[r];

    asm volatile("s_wait_dscnt 0x0" ::: "memory");  // P write -> P read (same wave)

    // O += P V : P is A [16 x 64keys], lVT is B [n=d][k=key].
    Frag pf[2];
#pragma unroll
    for (int ks = 0; ks < 2; ++ks) lds_frag(lP[w], 0, FS, ks * 32, pf[ks]);
#pragma unroll
    for (int nt = 0; nt < 4; ++nt)
#pragma unroll
      for (int ks = 0; ks < 2; ++ks) {
        Frag bvf;
        lds_frag(lVT, nt * 16, FS, ks * 32, bvf);
        o[nt] = wmma_bf16(pf[ks], bvf, o[nt]);
      }
  }

  // Final normalize + write y in [B,T,C] layout.
#pragma unroll
  for (int nt = 0; nt < 4; ++nt)
#pragma unroll
    for (int r = 0; r < 8; ++r) {
      const int qi = q0 + r + 8 * hi;
      const float val = o[nt][r] / lrow[r];
      y[((size_t)(b * SEQ) + qi) * N_EMBD + h * HEAD_D + nt * 16 + m] = f2bf(val);
    }
}

// ---------------------------------------------------------------------------
// Host-side launch
// ---------------------------------------------------------------------------
extern "C" void kernel_launch(void* const* d_in, const int* in_sizes, int n_in,
                              void* d_out, int out_size, void* d_ws, size_t ws_size,
                              hipStream_t stream) {
  (void)in_sizes; (void)n_in; (void)out_size; (void)ws_size;

  const float* xt   = (const float*)d_in[0];
  const float* xe   = (const float*)d_in[1];
  const float* ln1w = (const float*)d_in[2];
  const float* ln1b = (const float*)d_in[3];
  const float* ln2w = (const float*)d_in[4];
  const float* ln2b = (const float*)d_in[5];
  const float* Wqkv = (const float*)d_in[6];
  const float* bqkv = (const float*)d_in[7];
  const float* Wap  = (const float*)d_in[8];
  const float* bap  = (const float*)d_in[9];
  const float* Wfc  = (const float*)d_in[10];
  const float* bfc  = (const float*)d_in[11];
  const float* Wmp  = (const float*)d_in[12];
  const float* bmp  = (const float*)d_in[13];

  float* xt_out = (float*)d_out;                       // [B,T,C]
  float* xe_out = (float*)d_out + (size_t)MTOK * N_EMBD;

  // Workspace layout. h reuses the qkv+y region (both dead by then).
  char* ws = (char*)d_ws;
  const size_t SZ_BTC = (size_t)MTOK * N_EMBD * 2;     // 16 MB
  uint16_t* xtb  = (uint16_t*)(ws + 0);
  uint16_t* xn   = (uint16_t*)(ws + SZ_BTC);
  uint16_t* qkvb = (uint16_t*)(ws + 2 * SZ_BTC);                 // 48 MB
  uint16_t* yb   = (uint16_t*)(ws + 2 * SZ_BTC + 3 * SZ_BTC);    // 16 MB
  uint16_t* hb   = (uint16_t*)(ws + 2 * SZ_BTC);                 // 64 MB (overlap)
  char* wbase = ws + 6 * SZ_BTC;
  uint16_t* wqkv = (uint16_t*)(wbase);
  uint16_t* wap  = (uint16_t*)(wbase + (size_t)C3 * N_EMBD * 2);
  uint16_t* wfc  = (uint16_t*)(wbase + (size_t)C3 * N_EMBD * 2 + (size_t)N_EMBD * N_EMBD * 2);
  uint16_t* wmp  = (uint16_t*)(wbase + (size_t)C3 * N_EMBD * 2 + (size_t)N_EMBD * N_EMBD * 2 +
                               (size_t)C4 * N_EMBD * 2);

  auto cvt = [&](const float* s, uint16_t* d, int n) {
    k_f32_to_bf16<<<(n + 255) / 256, 256, 0, stream>>>(s, d, n);
  };
  cvt(Wqkv, wqkv, C3 * N_EMBD);
  cvt(Wap,  wap,  N_EMBD * N_EMBD);
  cvt(Wfc,  wfc,  C4 * N_EMBD);
  cvt(Wmp,  wmp,  N_EMBD * C4);

  // LN1 over xt+xe (also emits bf16(xt) for V-modulation).
  k_add_ln_bf16<<<MTOK, 256, 0, stream>>>(xt, xe, ln1w, ln1b, xn, xtb);

  // QKV GEMM: [8192,1024] x [3072,1024]^T -> bf16 [8192,3072]
  k_gemm_bf16<0><<<dim3(MTOK / 128, C3 / 128), 128, 0, stream>>>(
      xn, wqkv, bqkv, nullptr, qkvb, MTOK, C3, N_EMBD);

  // Flash attention -> y bf16 [8192,1024]
  k_flash_attn<<<dim3(SEQ / 128, N_HEADS, BATCH), 256, 0, stream>>>(qkvb, xtb, yb);

  // Attn proj + residual: xt_new = xt + y @ Wap^T + b  -> f32 d_out[0]
  k_gemm_bf16<2><<<dim3(MTOK / 128, N_EMBD / 128), 128, 0, stream>>>(
      yb, wap, bap, xt, xt_out, MTOK, N_EMBD, N_EMBD);

  // LN2 over xt_new + xe.
  k_add_ln_bf16<<<MTOK, 256, 0, stream>>>(xt_out, xe, ln2w, ln2b, xn, nullptr);

  // FC + GELU: [8192,1024] x [4096,1024]^T -> bf16 [8192,4096]
  k_gemm_bf16<1><<<dim3(MTOK / 128, C4 / 128), 128, 0, stream>>>(
      xn, wfc, bfc, nullptr, hb, MTOK, C4, N_EMBD);

  // MLP proj + residual: xe_new = xe + h @ Wmp^T + b -> f32 d_out[1]
  k_gemm_bf16<2><<<dim3(MTOK / 128, N_EMBD / 128), 128, 0, stream>>>(
      hb, wmp, bmp, xe, xe_out, MTOK, N_EMBD, C4);
}
